// DependencyParsingNetwork_70265664963201
// MI455X (gfx1250) — matrix operand order repaired and checked by reading, not compiled, not run on hardware
//
#include <hip/hip_runtime.h>
#include <hip/hip_bf16.h>

// ---------------------------------------------------------------------------
// Problem constants
// ---------------------------------------------------------------------------
#define TT   2048      // sequence length
#define HH   256       // hidden
#define DD   256       // embedding dim
#define G4H  1024      // 4*H

typedef __attribute__((ext_vector_type(16))) __bf16 v16bf;
typedef __attribute__((ext_vector_type(8)))  __bf16 v8bf;
typedef __attribute__((ext_vector_type(8)))  float  v8f;

// round-to-nearest-even f32 -> bf16 bits
__device__ __forceinline__ unsigned short f2bf(float f) {
    unsigned u = __float_as_uint(f);
    unsigned r = u + 0x7fffu + ((u >> 16) & 1u);
    return (unsigned short)(r >> 16);
}
__device__ __forceinline__ float bf_lo(unsigned w) { return __uint_as_float(w << 16); }
__device__ __forceinline__ float bf_hi(unsigned w) { return __uint_as_float(w & 0xffff0000u); }
__device__ __forceinline__ float sigm(float x) { return 1.0f / (1.0f + __expf(-x)); }

// ---------------------------------------------------------------------------
// Small conversion kernels
// ---------------------------------------------------------------------------
__global__ void cvt_bf16_kernel(const float* __restrict__ src,
                                unsigned short* __restrict__ dst, int n) {
    for (int i = blockIdx.x * blockDim.x + threadIdx.x; i < n;
         i += gridDim.x * blockDim.x)
        dst[i] = f2bf(src[i]);
}

// Whh [1024][256] f32 row-major -> WT [128][1024] u32; WT[k2][r] packs
// (Whh[r][2k2], Whh[r][2k2+1]) as bf16 pair.  Coalesced reads in the scan.
__global__ void pack_whh_kernel(const float* __restrict__ W,
                                unsigned* __restrict__ dst) {
    int idx = blockIdx.x * blockDim.x + threadIdx.x;   // 128*1024 threads
    int k2 = idx >> 10, r = idx & 1023;
    float w0 = W[r * HH + 2 * k2];
    float w1 = W[r * HH + 2 * k2 + 1];
    dst[idx] = (unsigned)f2bf(w0) | ((unsigned)f2bf(w1) << 16);
}

// embedding gather + bf16 convert:  xe[t][d] = bf16(E[word[t]][d])
__global__ void embed_kernel(const int* __restrict__ wi,
                             const float* __restrict__ E,
                             unsigned short* __restrict__ xe) {
    int idx = blockIdx.x * blockDim.x + threadIdx.x;   // 2048*256 threads
    int t = idx >> 8, d = idx & 255;
    xe[idx] = f2bf(E[(long)wi[t] * DD + d]);
}

// ---------------------------------------------------------------------------
// WMMA GEMM:  out[m][n] = sum_k A[m][k]*W[n][k] + bias[n]
//   A: [M][K] bf16 row-major, W: [1024][K] bf16 row-major (so B[k][n]=W[n][k])
//   grid = (M/16, 8, 2 dirs); block = 256 (8 waves, one 16x16 n-tile each)
// ---------------------------------------------------------------------------
__global__ void gemm_wmma_kernel(const unsigned short* __restrict__ Abf,
                                 const unsigned short* __restrict__ WfBf,
                                 const unsigned short* __restrict__ WbBf,
                                 const float* __restrict__ biasF,
                                 const float* __restrict__ biasB,
                                 float* __restrict__ outF,
                                 float* __restrict__ outB,
                                 int K) {
    const unsigned short* Wsel = blockIdx.z ? WbBf : WfBf;
    const float* bias = blockIdx.z ? biasB : biasF;
    float* out = blockIdx.z ? outB : outF;

    const int lane = threadIdx.x & 31;
    const int wave = threadIdx.x >> 5;
    const int m0 = blockIdx.x * 16;
    const int n0 = (blockIdx.y * 8 + wave) * 16;
    const int half = lane >> 4;        // 0: lanes 0-15, 1: lanes 16-31
    const int l15 = lane & 15;

    const __bf16* A = (const __bf16*)Abf;
    const __bf16* W = (const __bf16*)Wsel;

    const int mA = m0 + l15;           // A row this lane supplies
    const int nB = n0 + l15;           // B column this lane supplies

    v8f c = {};
    for (int k0 = 0; k0 < K; k0 += 32) {
        // A fragment 16x32 bf16: lane half selects K base 0/16,
        // elements 0..15 = K kA..kA+15 (contiguous, matches B)
        const int kA = k0 + half * 16;
        v8bf alo = *(const v8bf*)(A + (long)mA * K + kA);
        v8bf ahi = *(const v8bf*)(A + (long)mA * K + kA + 8);
        v16bf a;
#pragma unroll
        for (int i = 0; i < 8; ++i) { a[i] = alo[i]; a[i + 8] = ahi[i]; }

        // B fragment 32x16 bf16: lane = column, 16 consecutive K values
        const int kB = k0 + half * 16;
        v16bf b = *(const v16bf*)(W + (long)nB * K + kB);

        // gfx1250-specific prefetch of next B tile (global_prefetch_b8)
        if (k0 + 32 < K)
            __builtin_prefetch(W + (long)nB * K + kB + 32, 0, 1);

        c = __builtin_amdgcn_wmma_f32_16x16x32_bf16(
                false, a, false, b, (short)0, c, false, false);
    }

    const float bv = bias[nB];
    const int rbase = m0 + (half ? 8 : 0);
#pragma unroll
    for (int r = 0; r < 8; ++r)
        out[(long)(rbase + r) * G4H + nB] = c[r] + bv;
}

// ---------------------------------------------------------------------------
// Sequential LSTM scan for one layer, both directions (grid.x = 2).
// One 1024-thread workgroup per direction; thread r owns gate row r.
//   pre : [T][1024] f32 (input projection, bias already added)
//   WT  : [128][1024] u32 packed-transposed Whh (bf16 pairs)
//   hout: [T][256] f32 hidden states
//   hcat: [T][512] bf16, this direction writes columns [colOff, colOff+256)
// ---------------------------------------------------------------------------
__global__ void lstm_scan_kernel(const float* __restrict__ preF,
                                 const float* __restrict__ preB,
                                 const unsigned* __restrict__ WTf,
                                 const unsigned* __restrict__ WTb,
                                 float* __restrict__ hFout,
                                 float* __restrict__ hBout,
                                 unsigned short* __restrict__ hcat) {
    const int dir = blockIdx.x;                 // 0 = fwd, 1 = bwd
    const float* pre = dir ? preB : preF;
    const unsigned* WT = dir ? WTb : WTf;
    float* hout = dir ? hBout : hFout;
    const int colOff = dir * HH;
    const int tid = threadIdx.x;                // 0..1023

    __shared__ float hbuf[HH];                  // current hidden state
    __shared__ float gbuf[G4H];                 // gate pre-activations

    if (tid < HH) hbuf[tid] = 0.0f;
    float cst = 0.0f;                           // cell state (threads < 256)
    __syncthreads();

    const float2* hb2 = (const float2*)hbuf;

    for (int s = 0; s < TT; ++s) {
        const int t = dir ? (TT - 1 - s) : s;

        // GEMV: acc = pre[t][tid] + sum_k Whh[tid][k] * h[k]
        float acc = pre[(long)t * G4H + tid];
#pragma unroll 8
        for (int k2 = 0; k2 < HH / 2; ++k2) {
            unsigned w = WT[k2 * G4H + tid];    // coalesced, L2-resident
            float2 hh = hb2[k2];                // LDS broadcast
            acc += bf_lo(w) * hh.x + bf_hi(w) * hh.y;
        }
        gbuf[tid] = acc;
        __syncthreads();

        if (tid < HH) {
            const float gi = gbuf[tid];
            const float gf = gbuf[HH + tid];
            const float gg = gbuf[2 * HH + tid];
            const float go = gbuf[3 * HH + tid];
            cst = sigm(gf) * cst + sigm(gi) * tanhf(gg);
            const float h = sigm(go) * tanhf(cst);
            hbuf[tid] = h;
            hout[(long)t * HH + tid] = h;
            hcat[(long)t * (2 * HH) + colOff + tid] = f2bf(h);
        }
        __syncthreads();
    }
}

// ---------------------------------------------------------------------------
// Head/dep scores: s_head[t] = h1cat[t] . Wm[0:512], s_dep[t] = . Wm[512:1024]
// One block (256 threads) per timestep.
// ---------------------------------------------------------------------------
__global__ void heads_kernel(const float* __restrict__ h1f,
                             const float* __restrict__ h1b,
                             const float* __restrict__ Wm,
                             float* __restrict__ sh,
                             float* __restrict__ sd) {
    const int t = blockIdx.x;
    const int j = threadIdx.x;                  // 0..255
    const float hf = h1f[(long)t * HH + j];
    const float hb = h1b[(long)t * HH + j];
    __shared__ float rh[HH], rd[HH];
    rh[j] = hf * Wm[j]       + hb * Wm[HH + j];
    rd[j] = hf * Wm[512 + j] + hb * Wm[512 + HH + j];
    __syncthreads();
    for (int s = HH / 2; s > 0; s >>= 1) {
        if (j < s) { rh[j] += rh[j + s]; rd[j] += rd[j + s]; }
        __syncthreads();
    }
    if (j == 0) { sh[t] = rh[0]; sd[t] = rd[0]; }
}

// scores[i][j] = (j > i) ? tanh(s_head[i] + s_dep[j] + bm) : 0
__global__ void scores_kernel(const float* __restrict__ sh,
                              const float* __restrict__ sd,
                              const float* __restrict__ bm,
                              float* __restrict__ out) {
    const int idx = blockIdx.x * blockDim.x + threadIdx.x;   // 4M threads
    const int i = idx >> 11;
    const int j = idx & (TT - 1);
    out[idx] = (j > i) ? tanhf(sh[i] + sd[j] + bm[0]) : 0.0f;
}

// ---------------------------------------------------------------------------
// Host-side orchestration
// ---------------------------------------------------------------------------
extern "C" void kernel_launch(void* const* d_in, const int* in_sizes, int n_in,
                              void* d_out, int out_size, void* d_ws, size_t ws_size,
                              hipStream_t stream) {
    const int*   word_idx = (const int*)  d_in[0];
    const float* E        = (const float*)d_in[1];
    const float* Wih0f    = (const float*)d_in[2];
    const float* Whh0f    = (const float*)d_in[3];
    const float* b0f      = (const float*)d_in[4];
    const float* Wih0b    = (const float*)d_in[5];
    const float* Whh0b    = (const float*)d_in[6];
    const float* b0b      = (const float*)d_in[7];
    const float* Wih1f    = (const float*)d_in[8];
    const float* Whh1f    = (const float*)d_in[9];
    const float* b1f      = (const float*)d_in[10];
    const float* Wih1b    = (const float*)d_in[11];
    const float* Whh1b    = (const float*)d_in[12];
    const float* b1b      = (const float*)d_in[13];
    const float* Wm       = (const float*)d_in[14];
    const float* bm       = (const float*)d_in[15];
    float* out = (float*)d_out;

    char* ws = (char*)d_ws;
    // all sizes are multiples of 256B -> natural alignment
    const size_t OXE  = 0;                          // 2048*256 bf16   = 1 MB
    const size_t OW0F = OXE  + 1048576;             // 1024*256 bf16   = 512 KB
    const size_t OW0B = OW0F + 524288;
    const size_t OW1F = OW0B + 524288;              // 1024*512 bf16   = 1 MB
    const size_t OW1B = OW1F + 1048576;
    const size_t OT0F = OW1B + 1048576;             // 128*1024 u32    = 512 KB
    const size_t OT0B = OT0F + 524288;
    const size_t OT1F = OT0B + 524288;
    const size_t OT1B = OT1F + 524288;
    const size_t OPRF = OT1B + 524288;              // 2048*1024 f32   = 8 MB
    const size_t OPRB = OPRF + 8388608;
    const size_t OHC  = OPRB + 8388608;             // 2048*512 bf16   = 2 MB
    const size_t OH1F = OHC  + 2097152;             // 2048*256 f32    = 2 MB
    const size_t OH1B = OH1F + 2097152;
    const size_t OSH  = OH1B + 2097152;             // 2048 f32
    const size_t OSD  = OSH  + 8192;

    unsigned short* xe   = (unsigned short*)(ws + OXE);
    unsigned short* W0f  = (unsigned short*)(ws + OW0F);
    unsigned short* W0b  = (unsigned short*)(ws + OW0B);
    unsigned short* W1f  = (unsigned short*)(ws + OW1F);
    unsigned short* W1b  = (unsigned short*)(ws + OW1B);
    unsigned*       WT0f = (unsigned*)(ws + OT0F);
    unsigned*       WT0b = (unsigned*)(ws + OT0B);
    unsigned*       WT1f = (unsigned*)(ws + OT1F);
    unsigned*       WT1b = (unsigned*)(ws + OT1B);
    float*          preF = (float*)(ws + OPRF);
    float*          preB = (float*)(ws + OPRB);
    unsigned short* hcat = (unsigned short*)(ws + OHC);
    float*          h1f  = (float*)(ws + OH1F);
    float*          h1b  = (float*)(ws + OH1B);
    float*          sh   = (float*)(ws + OSH);
    float*          sd   = (float*)(ws + OSD);

    // 1) weight conversion / packing
    cvt_bf16_kernel<<<1024, 256, 0, stream>>>(Wih0f, W0f, G4H * DD);
    cvt_bf16_kernel<<<1024, 256, 0, stream>>>(Wih0b, W0b, G4H * DD);
    cvt_bf16_kernel<<<2048, 256, 0, stream>>>(Wih1f, W1f, G4H * 2 * HH);
    cvt_bf16_kernel<<<2048, 256, 0, stream>>>(Wih1b, W1b, G4H * 2 * HH);
    pack_whh_kernel<<<512, 256, 0, stream>>>(Whh0f, WT0f);
    pack_whh_kernel<<<512, 256, 0, stream>>>(Whh0b, WT0b);
    pack_whh_kernel<<<512, 256, 0, stream>>>(Whh1f, WT1f);
    pack_whh_kernel<<<512, 256, 0, stream>>>(Whh1b, WT1b);

    // 2) embedding gather -> bf16
    embed_kernel<<<2048, 256, 0, stream>>>(word_idx, E, xe);

    // 3) layer-0 input projections (WMMA), both directions
    gemm_wmma_kernel<<<dim3(TT / 16, 8, 2), 256, 0, stream>>>(
        xe, W0f, W0b, b0f, b0b, preF, preB, DD);

    // 4) layer-0 recurrence (fwd+bwd concurrently); f32 h into h1f/h1b (temp)
    lstm_scan_kernel<<<2, 1024, 0, stream>>>(preF, preB, WT0f, WT0b,
                                             h1f, h1b, hcat);

    // 5) layer-1 input projections from concatenated bf16 hidden states
    gemm_wmma_kernel<<<dim3(TT / 16, 8, 2), 256, 0, stream>>>(
        hcat, W1f, W1b, b1f, b1b, preF, preB, 2 * HH);

    // 6) layer-1 recurrence (hcat clobber is harmless; h1f/h1b now final)
    lstm_scan_kernel<<<2, 1024, 0, stream>>>(preF, preB, WT1f, WT1b,
                                             h1f, h1b, hcat);

    // 7) head/dep projections and masked tanh score matrix
    heads_kernel<<<TT, 256, 0, stream>>>(h1f, h1b, Wm, sh, sd);
    scores_kernel<<<(TT * TT) / 256, 256, 0, stream>>>(sh, sd, bm, out);
}
